// Attention_50096498541174
// MI455X (gfx1250) — matrix operand order repaired and checked by reading, not compiled, not run on hardware
//
#include <hip/hip_runtime.h>

typedef __bf16 bf16;
typedef __attribute__((ext_vector_type(16))) __bf16       v16bf;
typedef __attribute__((ext_vector_type(8)))  float        v8f;
typedef __attribute__((ext_vector_type(4)))  unsigned int v4u;
typedef __attribute__((ext_vector_type(4)))  int          i4;

#define B_ 32
#define T_ 128
#define S_ 512
#define H_ 1024
#define NEGV (-1e9f)

#if defined(__has_builtin)
#if __has_builtin(__builtin_amdgcn_global_load_async_to_lds_b128) && \
    __has_builtin(__builtin_amdgcn_s_wait_asynccnt)
#define HAVE_ASYNC_LDS 1
#endif
#endif

#ifdef HAVE_ASYNC_LDS
typedef __attribute__((address_space(1))) i4 glb_i4;   // global-space int4
typedef __attribute__((address_space(3))) i4 lds_i4;   // LDS-space int4
#endif

union FragU { v16bf v; v4u u[2]; };

// A-fragment (16x32 bf16) from a row-major tile (row stride in elements).
// Lane r=lane&15 holds row r; half=lane>>4 selects K bases {8*half, 16+8*half}.
__device__ __forceinline__ v16bf load_a(const bf16* tile, size_t stride, int lane) {
    const int r = lane & 15, half = lane >> 4;
    const bf16* p = tile + (size_t)r * stride + half * 8;
    FragU f;
    f.u[0] = *(const v4u*)(p);        // K = 8*half .. +7
    f.u[1] = *(const v4u*)(p + 16);   // K = 16+8*half .. +7
    return f.v;
}

// B-fragment (32x16 bf16) from Bt storage: Bt[n][k] row-major (k contiguous).
__device__ __forceinline__ v16bf load_b(const bf16* bt, size_t stride, int lane) {
    const int n = lane & 15, koff = (lane >> 4) * 8;
    const bf16* p = bt + (size_t)n * stride + koff;
    FragU f;
    f.u[0] = *(const v4u*)(p);        // K = koff .. +7
    f.u[1] = *(const v4u*)(p + 16);   // K = 16+koff .. +7
    return f.v;
}

__device__ __forceinline__ v8f wmma_bf16(v16bf a, v16bf b, v8f c) {
    return __builtin_amdgcn_wmma_f32_16x16x32_bf16(
        false, a, false, b, (short)0, c, false, false);
}

__device__ __forceinline__ v8f vzero8() {
    v8f z = {0.f, 0.f, 0.f, 0.f, 0.f, 0.f, 0.f, 0.f};
    return z;
}

// Cheap tanh: exact formula via exp, clamped to avoid inf/inf (tanh(15)==1 in fp32).
__device__ __forceinline__ float fast_tanh(float x) {
    x = fminf(fmaxf(x, -15.f), 15.f);
    const float e = __expf(2.f * x);
    return (e - 1.f) / (e + 1.f);
}

// ---------------- Kernel 1: fp32 -> bf16 ----------------
__global__ void cvt_kernel(const float* __restrict__ src, bf16* __restrict__ dst, int n) {
    int i = blockIdx.x * blockDim.x + threadIdx.x;
    int stride = gridDim.x * blockDim.x;
    for (; i < n; i += stride) dst[i] = (bf16)src[i];
}

// ---------------- Kernel 1b: encT[b][h][s] = (bf16)enc[b][s][h] ----------------
__global__ void __launch_bounds__(256) transpose_kernel(const float* __restrict__ enc,
                                                        bf16* __restrict__ encT) {
    __shared__ bf16 tile[64][65];
    const int tid = threadIdx.x;
    const int b = blockIdx.z;
    const int s0 = blockIdx.x * 64, h0 = blockIdx.y * 64;

    const float* src = enc + ((size_t)b * S_ + s0) * H_ + h0;
    const int ss = tid >> 2, hh0 = (tid & 3) * 16;
#pragma unroll
    for (int j = 0; j < 16; ++j)
        tile[hh0 + j][ss] = (bf16)src[(size_t)ss * H_ + hh0 + j];
    __syncthreads();

    bf16* dst = encT + ((size_t)b * H_ + h0) * S_ + s0;
    const int hh = tid >> 2, ss0 = (tid & 3) * 16;
#pragma unroll
    for (int j = 0; j < 16; ++j)
        dst[(size_t)hh * S_ + ss0 + j] = tile[hh][ss0 + j];
}

// ---------------- Kernel 2: q_proj = Q * W_in^T ----------------
// Software-pipelined: next k-step's A + 8 B fragments are loaded before the
// current WMMA burst, with a peeled final step. Barrier/LDS-free.
__global__ void __launch_bounds__(256, 1) qproj_kernel(const bf16* __restrict__ Q,
                                                       const bf16* __restrict__ Win,
                                                       bf16* __restrict__ QP) {
    const int tid = threadIdx.x, lane = tid & 31, wave = tid >> 5;
    const int m0 = blockIdx.x * 16;
    const int n0 = wave * 128;
    const bf16* Ab = Q + (size_t)m0 * H_;

    v8f acc[8];
#pragma unroll
    for (int i = 0; i < 8; ++i) acc[i] = vzero8();

    v16bf a_cur = load_a(Ab, H_, lane);
    v16bf b_cur[8];
#pragma unroll
    for (int nt = 0; nt < 8; ++nt)
        b_cur[nt] = load_b(Win + (size_t)(n0 + nt * 16) * H_, H_, lane);

#pragma unroll 2
    for (int k0 = 0; k0 < H_ - 32; k0 += 32) {
        v16bf a_nxt = load_a(Ab + k0 + 32, H_, lane);
        v16bf b_nxt[8];
#pragma unroll
        for (int nt = 0; nt < 8; ++nt)
            b_nxt[nt] = load_b(Win + (size_t)(n0 + nt * 16) * H_ + k0 + 32, H_, lane);
#pragma unroll
        for (int nt = 0; nt < 8; ++nt)
            acc[nt] = wmma_bf16(a_cur, b_cur[nt], acc[nt]);
        a_cur = a_nxt;
#pragma unroll
        for (int nt = 0; nt < 8; ++nt) b_cur[nt] = b_nxt[nt];
    }
#pragma unroll
    for (int nt = 0; nt < 8; ++nt)
        acc[nt] = wmma_bf16(a_cur, b_cur[nt], acc[nt]);

    const int half = lane >> 4, n = lane & 15;
#pragma unroll
    for (int nt = 0; nt < 8; ++nt)
#pragma unroll
        for (int i = 0; i < 8; ++i)
            QP[(size_t)(m0 + i + 8 * half) * H_ + n0 + nt * 16 + n] = (bf16)acc[nt][i];
}

// ---------------- Kernel 3: fused score/softmax/context ----------------
// One workgroup per (b, 16-row t tile). 256 threads = 8 waves.
// LDS: qp[16][1024] bf16 (32KB) | sc[16][512] f32 (32KB) | p[16][512] bf16 (16KB) = 80KB.
__global__ void __launch_bounds__(256, 1) attn_kernel(const bf16* __restrict__ QP,
                                                      const bf16* __restrict__ ENC,
                                                      const bf16* __restrict__ ENCT,
                                                      const int* __restrict__ lens,
                                                      bf16* __restrict__ CTX) {
    extern __shared__ char smem[];
    bf16*  qp   = (bf16*)(smem);
    float* sc   = (float*)(smem + 32768);
    bf16*  pbuf = (bf16*)(smem + 65536);

    const int tid = threadIdx.x, lane = tid & 31, wave = tid >> 5;
    const int half = lane >> 4, n = lane & 15;
    const int b  = blockIdx.x >> 3;
    const int t0 = (blockIdx.x & 7) * 16;
    const bf16* qpg   = QP   + (size_t)(b * T_ + t0) * H_;
    const bf16* encg  = ENC  + (size_t)b * S_ * H_;
    const bf16* encTg = ENCT + (size_t)b * H_ * S_;

    // Stage the 16x1024 q_proj tile via the CDNA5 async copy path if present.
#ifdef HAVE_ASYNC_LDS
    for (int i = tid; i < 16 * H_ / 8; i += 256) {
        __builtin_amdgcn_global_load_async_to_lds_b128(
            (glb_i4*)(qpg + (size_t)i * 8),
            (lds_i4*)(qp + (size_t)i * 8),
            /*offset=*/0, /*cpol=*/0);
    }
    __builtin_amdgcn_s_wait_asynccnt(0);
#else
    {
        const v4u* src = (const v4u*)qpg;
        v4u* dst = (v4u*)qp;
        for (int i = tid; i < 16 * H_ / 8; i += 256) dst[i] = src[i];
    }
#endif
    __syncthreads();

    // ---- Scores: wave owns 4 s-tiles; pipelined B-frag loads from global.
    {
        v8f sacc[4];
#pragma unroll
        for (int st = 0; st < 4; ++st) sacc[st] = vzero8();

        v16bf b_cur[4];
#pragma unroll
        for (int st = 0; st < 4; ++st)
            b_cur[st] = load_b(encg + (size_t)((wave * 4 + st) * 16) * H_, H_, lane);

#pragma unroll 2
        for (int k0 = 0; k0 < H_ - 32; k0 += 32) {
            v16bf b_nxt[4];
#pragma unroll
            for (int st = 0; st < 4; ++st)
                b_nxt[st] = load_b(encg + (size_t)((wave * 4 + st) * 16) * H_ + k0 + 32, H_, lane);
            v16bf a = load_a(qp + k0, H_, lane);
#pragma unroll
            for (int st = 0; st < 4; ++st)
                sacc[st] = wmma_bf16(a, b_cur[st], sacc[st]);
#pragma unroll
            for (int st = 0; st < 4; ++st) b_cur[st] = b_nxt[st];
        }
        {
            v16bf a = load_a(qp + H_ - 32, H_, lane);
#pragma unroll
            for (int st = 0; st < 4; ++st)
                sacc[st] = wmma_bf16(a, b_cur[st], sacc[st]);
        }
#pragma unroll
        for (int st = 0; st < 4; ++st) {
            const int s0 = (wave * 4 + st) * 16;
#pragma unroll
            for (int i = 0; i < 8; ++i)
                sc[(i + 8 * half) * S_ + s0 + n] = sacc[st][i];
        }
    }
    __syncthreads();

    // ---- Masked softmax: 16 threads per row (half-wave), 32 cols each.
    {
        const int r = tid >> 4, c = tid & 15;
        const int len = lens[b];
        const float* row = sc + r * S_ + c * 32;
        float vals[32];
        float mx = -3.4e38f;
#pragma unroll
        for (int j = 0; j < 32; ++j) {
            const int s = c * 32 + j;
            const float v = (s < len) ? row[j] : NEGV;
            vals[j] = v;
            mx = fmaxf(mx, v);
        }
#pragma unroll
        for (int m = 8; m >= 1; m >>= 1) mx = fmaxf(mx, __shfl_xor(mx, m, 32));
        float sum = 0.f;
#pragma unroll
        for (int j = 0; j < 32; ++j) { float e = __expf(vals[j] - mx); vals[j] = e; sum += e; }
#pragma unroll
        for (int m = 8; m >= 1; m >>= 1) sum += __shfl_xor(sum, m, 32);
        const float inv = 1.f / sum;
        bf16* prow = pbuf + r * S_ + c * 32;
#pragma unroll
        for (int j = 0; j < 32; ++j) prow[j] = (bf16)(vals[j] * inv);
    }
    __syncthreads();

    // ---- Context: c[16,1024] = p[16,512] @ enc[512,1024], B from encT rows.
    v8f acc[8];
#pragma unroll
    for (int i = 0; i < 8; ++i) acc[i] = vzero8();

    v16bf a_cur = load_a(pbuf, S_, lane);
    v16bf b_cur[8];
#pragma unroll
    for (int ht = 0; ht < 8; ++ht)
        b_cur[ht] = load_b(encTg + (size_t)(wave * 128 + ht * 16) * S_, S_, lane);

#pragma unroll 2
    for (int k0 = 0; k0 < S_ - 32; k0 += 32) {
        v16bf a_nxt = load_a(pbuf + k0 + 32, S_, lane);
        v16bf b_nxt[8];
#pragma unroll
        for (int ht = 0; ht < 8; ++ht)
            b_nxt[ht] = load_b(encTg + (size_t)(wave * 128 + ht * 16) * S_ + k0 + 32, S_, lane);
#pragma unroll
        for (int ht = 0; ht < 8; ++ht)
            acc[ht] = wmma_bf16(a_cur, b_cur[ht], acc[ht]);
        a_cur = a_nxt;
#pragma unroll
        for (int ht = 0; ht < 8; ++ht) b_cur[ht] = b_nxt[ht];
    }
#pragma unroll
    for (int ht = 0; ht < 8; ++ht)
        acc[ht] = wmma_bf16(a_cur, b_cur[ht], acc[ht]);

    bf16* cg = CTX + (size_t)(b * T_ + t0) * H_;
#pragma unroll
    for (int ht = 0; ht < 8; ++ht)
#pragma unroll
        for (int i = 0; i < 8; ++i)
            cg[(size_t)(i + 8 * half) * H_ + wave * 128 + ht * 16 + n] = (bf16)acc[ht][i];
}

// ---------------- Kernel 4: out = tanh(concat(Q,C) * W_out^T + b) ----------------
__global__ void __launch_bounds__(256, 1) out_kernel(const bf16* __restrict__ Q,
                                                     const bf16* __restrict__ CTX,
                                                     const bf16* __restrict__ Wout,
                                                     const float* __restrict__ bias,
                                                     float* __restrict__ out) {
    const int tid = threadIdx.x, lane = tid & 31, wave = tid >> 5;
    const int m0 = blockIdx.x * 16;
    const int n0 = wave * 128;

    auto aptr = [&](int k) -> const bf16* {
        return (k < H_) ? (Q + (size_t)m0 * H_ + k)
                        : (CTX + (size_t)m0 * H_ + (k - H_));
    };

    v8f acc[8];
#pragma unroll
    for (int i = 0; i < 8; ++i) acc[i] = vzero8();

    v16bf a_cur = load_a(aptr(0), H_, lane);
    v16bf b_cur[8];
#pragma unroll
    for (int nt = 0; nt < 8; ++nt)
        b_cur[nt] = load_b(Wout + (size_t)(n0 + nt * 16) * (2 * H_), 2 * H_, lane);

#pragma unroll 2
    for (int k0 = 0; k0 < 2 * H_ - 32; k0 += 32) {
        v16bf a_nxt = load_a(aptr(k0 + 32), H_, lane);
        v16bf b_nxt[8];
#pragma unroll
        for (int nt = 0; nt < 8; ++nt)
            b_nxt[nt] = load_b(Wout + (size_t)(n0 + nt * 16) * (2 * H_) + k0 + 32, 2 * H_, lane);
#pragma unroll
        for (int nt = 0; nt < 8; ++nt)
            acc[nt] = wmma_bf16(a_cur, b_cur[nt], acc[nt]);
        a_cur = a_nxt;
#pragma unroll
        for (int nt = 0; nt < 8; ++nt) b_cur[nt] = b_nxt[nt];
    }
#pragma unroll
    for (int nt = 0; nt < 8; ++nt)
        acc[nt] = wmma_bf16(a_cur, b_cur[nt], acc[nt]);

    const int half = lane >> 4, n = lane & 15;
#pragma unroll
    for (int nt = 0; nt < 8; ++nt) {
        const int h = n0 + nt * 16 + n;
        const float bb = bias[h];
#pragma unroll
        for (int i = 0; i < 8; ++i) {
            const int m = m0 + i + 8 * half;
            out[(size_t)m * H_ + h] = fast_tanh(acc[nt][i] + bb);
        }
    }
}

// ---------------- Host side ----------------
extern "C" void kernel_launch(void* const* d_in, const int* in_sizes, int n_in,
                              void* d_out, int out_size, void* d_ws, size_t ws_size,
                              hipStream_t stream) {
    const float* q    = (const float*)d_in[0];   // [32,128,1024]
    const float* enc  = (const float*)d_in[1];   // [32,512,1024]
    const int*   slen = (const int*)  d_in[2];   // [32]
    const float* win  = (const float*)d_in[3];   // [1024,1024]
    const float* wout = (const float*)d_in[4];   // [1024,2048]
    const float* bout = (const float*)d_in[5];   // [1024]
    float* out = (float*)d_out;

    char* ws = (char*)d_ws;
    bf16* qb    = (bf16*)(ws + 0);                        //  8 MB  [4096,1024]
    bf16* encb  = (bf16*)(ws + (size_t)8  * 1024 * 1024); // 32 MB [32,512,1024]
    bf16* encTb = (bf16*)(ws + (size_t)40 * 1024 * 1024); // 32 MB [32,1024,512]
    bf16* winb  = (bf16*)(ws + (size_t)72 * 1024 * 1024); //  2 MB
    bf16* woutb = (bf16*)(ws + (size_t)74 * 1024 * 1024); //  4 MB
    bf16* qpb   = (bf16*)(ws + (size_t)78 * 1024 * 1024); //  8 MB
    bf16* ctxb  = (bf16*)(ws + (size_t)86 * 1024 * 1024); //  8 MB  (total 94 MB)
    if (ws_size < (size_t)94 * 1024 * 1024) return;

    cvt_kernel<<<4096, 256, 0, stream>>>(q,    qb,    B_ * T_ * H_);
    cvt_kernel<<<8192, 256, 0, stream>>>(enc,  encb,  B_ * S_ * H_);
    cvt_kernel<<<1024, 256, 0, stream>>>(win,  winb,  H_ * H_);
    cvt_kernel<<<2048, 256, 0, stream>>>(wout, woutb, H_ * 2 * H_);
    transpose_kernel<<<dim3(S_ / 64, H_ / 64, B_), 256, 0, stream>>>(enc, encTb);

    qproj_kernel<<<256, 256, 0, stream>>>(qb, winb, qpb);
    attn_kernel<<<256, 256, 81920, stream>>>(qpb, encb, encTb, slen, ctxb);
    out_kernel<<<256, 256, 0, stream>>>(qb, ctxb, woutb, bout, out);
}
